// SeparableBilinConvUpsampling_73237782331581
// MI455X (gfx1250) — matrix operand order repaired and checked by reading, not compiled, not run on hardware
//
#include <hip/hip_runtime.h>

typedef _Float16 v16h __attribute__((ext_vector_type(16)));
typedef float    v8f  __attribute__((ext_vector_type(8)));

#define C_IN   256
#define C_OUT  128
#define IN_H   60
#define IN_W   80
#define OUT_W  160
#define SPB    19200                 // 120*160 pixels per batch image
#define N_TOT  (8 * SPB)             // 153600 spatial positions
#define PITCH  272                   // f16 row pitch: 544B = 17*32B -> aligned v16h

// ---------------------------------------------------------------------------
// Fully fused: bilinear upsample (align_corners) + 3x3 depthwise -> LDS (f16)
// -> pointwise GEMM via v_wmma_f32_16x16x32_f16 -> out.
// Block = 256 threads (8 waves) handles 32 consecutive pixels (one row
// segment: 160 % 32 == 0) for all 256 in / 128 out channels.
// Phase A: lane = pixel, wave sweeps 32 channels; per channel only the true
//   3x3 input footprint is loaded (9 loads) and mixed with per-pixel dense
//   row/col weight matrices RW/CW (channel-independent, built once).
// Phase B: wave w owns M subtile [16w,16w+16), two N-16 subtiles; B fragments
//   read 32B-aligned from LDS, A fragments from w_pw (f32 -> f16 on the fly).
// ---------------------------------------------------------------------------
__global__ __launch_bounds__(256)
void fused_updw_pw_wmma_kernel(const float* __restrict__ x,
                               const float* __restrict__ w_dw,
                               const float* __restrict__ w_pw,
                               float* __restrict__ out) {
    __shared__ _Float16 lds[32 * PITCH];

    const int tid  = threadIdx.x;
    const int lane = tid & 31;
    const int wv   = tid >> 5;
    const int n0   = blockIdx.x * 32;

    // ---------------- Phase A: upsample + depthwise -> LDS f16 -------------
    const int b    = n0 / SPB;
    const int s0   = n0 - b * SPB;
    const int y    = s0 / OUT_W;                 // uniform across block
    const int xpix = (s0 - y * OUT_W) + lane;    // lane's output column

    const float ry = 59.0f / 121.0f;             // (H-1)/(RES_H-1)
    const float rx = 79.0f / 161.0f;             // (W-1)/(RES_W-1)

    // 3x3 input footprint bases (clamped) + dense per-pixel mixing matrices
    const int r0 = (int)((float)y * ry);
    const int g0 = (int)((float)xpix * rx);
    int rowoff[3], coloff[3];
#pragma unroll
    for (int k = 0; k < 3; ++k) {
        const int rr = r0 + k;
        rowoff[k] = ((rr < IN_H - 1) ? rr : (IN_H - 1)) * IN_W;
        const int gg = g0 + k;
        coloff[k] = (gg < IN_W - 1) ? gg : (IN_W - 1);
    }
    float RW[3][3], CW[3][3];
#pragma unroll
    for (int i = 0; i < 3; ++i) {                 // row weights (wave-uniform)
        const float hy = (float)(y + i) * ry;
        const int   y0 = (int)hy;
        const float f  = hy - (float)y0;
        const int a0 = y0 - r0;
        const int a1 = ((y0 + 1 < IN_H - 1) ? y0 + 1 : (IN_H - 1)) - r0;
#pragma unroll
        for (int k = 0; k < 3; ++k)
            RW[i][k] = ((a0 == k) ? (1.0f - f) : 0.0f) + ((a1 == k) ? f : 0.0f);
    }
#pragma unroll
    for (int j = 0; j < 3; ++j) {                 // col weights (per lane)
        const float wx = (float)(xpix + j) * rx;
        const int   x0 = (int)wx;
        const float f  = wx - (float)x0;
        const int a0 = x0 - g0;
        const int a1 = ((x0 + 1 < IN_W - 1) ? x0 + 1 : (IN_W - 1)) - g0;
#pragma unroll
        for (int k = 0; k < 3; ++k)
            CW[j][k] = ((a0 == k) ? (1.0f - f) : 0.0f) + ((a1 == k) ? f : 0.0f);
    }

    const float* xb = x + (size_t)b * C_IN * (IN_H * IN_W);
    for (int cc = 0; cc < 32; ++cc) {
        const int c = wv * 32 + cc;
        const float* xc = xb + (size_t)c * (IN_H * IN_W);
        const float* wd = w_dw + c * 9;

        float v[3][3];
#pragma unroll
        for (int r = 0; r < 3; ++r)
#pragma unroll
            for (int g = 0; g < 3; ++g)
                v[r][g] = xc[rowoff[r] + coloff[g]];

        float h[3][3];                            // horizontal mix: 27 fma
#pragma unroll
        for (int r = 0; r < 3; ++r)
#pragma unroll
            for (int j = 0; j < 3; ++j)
                h[r][j] = CW[j][0] * v[r][0] + CW[j][1] * v[r][1] + CW[j][2] * v[r][2];

        float acc = 0.0f;                         // vertical mix + dw: 36 fma
#pragma unroll
        for (int i = 0; i < 3; ++i)
#pragma unroll
            for (int j = 0; j < 3; ++j) {
                const float u = RW[i][0] * h[0][j] + RW[i][1] * h[1][j] + RW[i][2] * h[2][j];
                acc += wd[i * 3 + j] * u;
            }

        lds[lane * PITCH + c] = (_Float16)acc;
    }
    __syncthreads();

    // ---------------- Phase B: pointwise GEMM via WMMA ---------------------
    const int l16  = lane & 15;
    const int half = lane >> 4;                   // 0: lanes 0-15, 1: 16-31
    const int m0   = wv * 16;

    // A fragments: w_pw row m0+l16, f32 -> f16, contiguous K layout:
    // VGPR v holds K=2v,2v+1; lane>=16: +16 (matches B)
    const float* wrow = w_pw + (size_t)(m0 + l16) * C_IN;
    const int koff = half * 16;
    v16h a[8];
#pragma unroll
    for (int ki = 0; ki < 8; ++ki) {
        const int kb = ki * 32 + koff;
#pragma unroll
        for (int v = 0; v < 8; ++v) {
            const int k = kb + 2 * v;
            a[ki][2 * v]     = (_Float16)wrow[k];
            a[ki][2 * v + 1] = (_Float16)wrow[k + 1];
        }
    }

    // B fragments from LDS: row N = l16 (+16 for 2nd subtile), 16 consecutive
    // K per lane; 544B row pitch keeps every access 32B-aligned.
    const _Float16* brow0 = lds + l16 * PITCH + half * 16;
    const _Float16* brow1 = lds + (16 + l16) * PITCH + half * 16;

    v8f c0 = {}, c1 = {};
#pragma unroll
    for (int ki = 0; ki < 8; ++ki) {
        const v16h b0 = *(const v16h*)(brow0 + ki * 32);
        const v16h b1 = *(const v16h*)(brow1 + ki * 32);
        c0 = __builtin_amdgcn_wmma_f32_16x16x32_f16(false, a[ki], false, b0,
                                                    (short)0, c0, false, false);
        c1 = __builtin_amdgcn_wmma_f32_16x16x32_f16(false, a[ki], false, b1,
                                                    (short)0, c1, false, false);
    }

    // C/D layout: VGPR r -> M = m0 + r + 8*half, N = n0 + l16 (+16)
    const int n_a = n0 + l16;                     // whole tile in one batch
    const int s   = n_a - b * SPB;
    float* obase = out + (size_t)b * C_OUT * SPB + s;
#pragma unroll
    for (int r = 0; r < 8; ++r) {
        const int m = m0 + r + half * 8;
        obase[(size_t)m * SPB]      = c0[r];
        obase[(size_t)m * SPB + 16] = c1[r];
    }
}

extern "C" void kernel_launch(void* const* d_in, const int* in_sizes, int n_in,
                              void* d_out, int out_size, void* d_ws, size_t ws_size,
                              hipStream_t stream) {
    const float* x    = (const float*)d_in[0];   // [8,256,60,80]
    const float* w_dw = (const float*)d_in[1];   // [256,1,3,3]
    const float* w_pw = (const float*)d_in[2];   // [128,256]
    float* out        = (float*)d_out;           // [8,128,120,160]

    (void)in_sizes; (void)n_in; (void)out_size; (void)d_ws; (void)ws_size;

    fused_updw_pw_wmma_kernel<<<N_TOT / 32, 256, 0, stream>>>(x, w_dw, w_pw, out);
}
